// GAT_47914655154530
// MI455X (gfx1250) — matrix operand order repaired and checked by reading, not compiled, not run on hardware
//
#include <hip/hip_runtime.h>
#include <hip/hip_bf16.h>

// ---------------------------------------------------------------------------
// GAT forward for MI455X (gfx1250, wave32).
// Dense node-feature GEMMs go through v_wmma_f32_16x16x32_bf16 (fp32 accum).
// Edge softmax/aggregation is HBM-bound -> single-pass kernels with native
// float atomics; segment-max via ordered-uint atomicMax.
// Problem sizes (fixed by the reference): N=50000, E=800000, F_IN=256,
// EMB=128, GF=32, G=256.  N%16==0, K%32==0, EMB%16==0 -> no GEMM predication,
// EXEC is all-ones inside WMMA kernels as the ISA requires.
// ---------------------------------------------------------------------------

#define NN      50000
#define EE_RAW  800000
#define EE_ALL  (EE_RAW + NN)     // + self loops
#define F_IN    256
#define EMB     128
#define GFT     32
#define GG      256
#define NEG_SLOPE 0.2f

typedef __attribute__((ext_vector_type(16))) __bf16 v16bf;
typedef __attribute__((ext_vector_type(8)))  float  v8f;

// ---- helpers --------------------------------------------------------------

__device__ __forceinline__ unsigned short f2bf(float f) {
  unsigned x = __float_as_uint(f);
  unsigned r = (x + 0x7FFFu + ((x >> 16) & 1u)) >> 16;  // round-nearest-even
  return (unsigned short)r;
}

// order-preserving float<->uint transform for atomicMax on floats
__device__ __forceinline__ unsigned encOrd(float v) {
  unsigned u = __float_as_uint(v);
  return (u & 0x80000000u) ? ~u : (u | 0x80000000u);
}
__device__ __forceinline__ float decOrd(unsigned u) {
  unsigned b = (u & 0x80000000u) ? (u & 0x7FFFFFFFu) : ~u;
  return __uint_as_float(b);
}
#define ENC_NEG_INF 0x007FFFFFu   // encOrd(-inf)

__device__ __forceinline__ void atomicMaxF(unsigned* addr, float v) {
  atomicMax(addr, encOrd(v));
}

// ---- conversion kernels ---------------------------------------------------

__global__ void k_f32_to_bf16(const float* __restrict__ in,
                              unsigned short* __restrict__ out, int n) {
  int i = blockIdx.x * blockDim.x + threadIdx.x;
  if (i < n) out[i] = f2bf(in[i]);
}

// W is [K x Ncol] row-major; write Wt bf16 as [Ncol x K] so B-fragment loads
// see consecutive K.
__global__ void k_transpose_bf16(const float* __restrict__ W,
                                 unsigned short* __restrict__ Wt,
                                 int K, int Ncol) {
  int i = blockIdx.x * blockDim.x + threadIdx.x;
  if (i >= K * Ncol) return;
  int k = i / Ncol, n = i % Ncol;
  Wt[n * K + k] = f2bf(W[i]);
}

__global__ void k_fill_u32(unsigned* __restrict__ p, unsigned v, int n) {
  int i = blockIdx.x * blockDim.x + threadIdx.x;
  if (i < n) p[i] = v;
}

// ---- WMMA GEMM: C[M x 128] = A[M x K](bf16) * Wt[128 x K](bf16) -----------
// Block = 256 threads = 8 waves; block b -> rows [16b,16b+16); wave w ->
// cols [16w,16w+16).  Requires M%16==0, K%32==0.

__global__ __launch_bounds__(256)
void k_gemm_bf16_wmma(const unsigned short* __restrict__ A,
                      const unsigned short* __restrict__ Bt,
                      float* __restrict__ C, int M, int K) {
  const int lane  = threadIdx.x & 31;
  const int wave  = threadIdx.x >> 5;
  const int m0    = blockIdx.x * 16;
  const int n0    = wave * 16;
  const int mrow  = m0 + (lane & 15);
  const int ncol  = n0 + (lane & 15);
  const int ahalf = (lane >> 4) * 8;    // upper half-wave: +8 in K
  const int bkb   = (lane >> 4) * 8;    // upper half-wave: +8 in K

  const unsigned* Arow = (const unsigned*)(A  + (size_t)mrow * K);
  const unsigned* Brow = (const unsigned*)(Bt + (size_t)ncol * K);

  v8f acc = {};
  for (int k0 = 0; k0 < K; k0 += 32) {
    union { v16bf v; unsigned u[8]; } a, b;
#pragma unroll
    for (int v = 0; v < 8; ++v) {
      int abase = (v < 4) ? (2 * v) : (16 + 2 * (v - 4));
      a.u[v] = Arow[(k0 + abase + ahalf) >> 1];
      b.u[v] = Brow[(k0 + abase + bkb) >> 1];
    }
    acc = __builtin_amdgcn_wmma_f32_16x16x32_bf16(
        false, a.v, false, b.v, (short)0, acc, false, false);
  }

  const int rbase = (lane >> 4) * 8;    // C/D: VGPR r -> M=r (+8 upper lanes)
#pragma unroll
  for (int r = 0; r < 8; ++r)
    C[(size_t)(m0 + rbase + r) * EMB + ncol] = acc[r];
}

// ---- per-node attention dots: s_src[n]=h[n]·a_src, s_dst[n]=h[n]·a_dst ----

__global__ __launch_bounds__(256)
void k_node_dots(const float* __restrict__ H, const float* __restrict__ as,
                 const float* __restrict__ ad, float* __restrict__ ss,
                 float* __restrict__ sd, int M) {
  int wave = (int)((blockIdx.x * blockDim.x + threadIdx.x) >> 5);
  int lane = threadIdx.x & 31;
  if (wave >= M) return;
  const float4* row = (const float4*)(H + (size_t)wave * EMB);
  const float4* A4  = (const float4*)as;
  const float4* D4  = (const float4*)ad;
  float4 h = row[lane], a = A4[lane], d = D4[lane];
  float vs = h.x * a.x + h.y * a.y + h.z * a.z + h.w * a.w;
  float vd = h.x * d.x + h.y * d.y + h.z * d.z + h.w * d.w;
  for (int off = 16; off; off >>= 1) {
    vs += __shfl_xor(vs, off);
    vd += __shfl_xor(vd, off);
  }
  if (lane == 0) { ss[wave] = vs; sd[wave] = vd; }
}

// ---- edge passes ----------------------------------------------------------

__global__ void k_edge_logits(const int* __restrict__ src,
                              const int* __restrict__ dst,
                              const float* __restrict__ ss,
                              const float* __restrict__ sd,
                              float* __restrict__ e, unsigned* __restrict__ m) {
  int i = blockIdx.x * blockDim.x + threadIdx.x;
  if (i >= EE_ALL) return;
  int s = (i < EE_RAW) ? src[i] : (i - EE_RAW);
  int d = (i < EE_RAW) ? dst[i] : (i - EE_RAW);
  float v = ss[s] + sd[d];
  v = (v > 0.f) ? v : NEG_SLOPE * v;      // leaky relu
  e[i] = v;
  atomicMaxF(&m[d], v);
}

__global__ void k_edge_exp(const int* __restrict__ dst,
                           const float* __restrict__ e,
                           const unsigned* __restrict__ m,
                           float* __restrict__ p, float* __restrict__ den) {
  int i = blockIdx.x * blockDim.x + threadIdx.x;
  if (i >= EE_ALL) return;
  int d = (i < EE_RAW) ? dst[i] : (i - EE_RAW);
  float pv = __expf(e[i] - decOrd(m[d]));  // self-loops => m[d] finite
  p[i] = pv;
  atomicAdd(&den[d], pv);
}

// one wave per edge: lane covers 4 features (32*4 = 128)
__global__ __launch_bounds__(256)
void k_edge_scatter(const int* __restrict__ src, const int* __restrict__ dst,
                    const float* __restrict__ p, const float* __restrict__ den,
                    const float* __restrict__ H, float* __restrict__ out) {
  int lane = threadIdx.x & 31;
  int e = (int)(blockIdx.x * (blockDim.x >> 5) + (threadIdx.x >> 5));
  if (e >= EE_ALL) return;
  int s = (e < EE_RAW) ? src[e] : (e - EE_RAW);
  int d = (e < EE_RAW) ? dst[e] : (e - EE_RAW);
  float alpha = p[e] / den[d];
  float4 h = ((const float4*)(H + (size_t)s * EMB))[lane];
  float* o = out + (size_t)d * EMB + lane * 4;
  atomicAdd(o + 0, alpha * h.x);
  atomicAdd(o + 1, alpha * h.y);
  atomicAdd(o + 2, alpha * h.z);
  atomicAdd(o + 3, alpha * h.w);
}

// ---- bias add (+ bf16 copy for next GEMM) ---------------------------------

__global__ void k_bias_bf16(float* __restrict__ H, const float* __restrict__ b,
                            unsigned short* __restrict__ Hb, int total) {
  int i = blockIdx.x * blockDim.x + threadIdx.x;
  if (i >= total) return;
  float v = H[i] + b[i & (EMB - 1)];
  H[i] = v;
  Hb[i] = f2bf(v);
}

// ---- pooling --------------------------------------------------------------

__global__ void k_pool(const float* __restrict__ H, const int* __restrict__ bat,
                       float* __restrict__ sum, unsigned* __restrict__ mx,
                       float* __restrict__ cnt) {
  int i = blockIdx.x * blockDim.x + threadIdx.x;
  if (i >= NN * EMB) return;
  int n = i >> 7, c = i & (EMB - 1);
  int g = bat[n];
  float v = H[i];
  atomicAdd(&sum[g * EMB + c], v);
  atomicMaxF(&mx[g * EMB + c], v);
  if (c == 0) atomicAdd(&cnt[g], 1.0f);
}

// ---- lin_g: [G,32] @ [32,128] + bg ---------------------------------------

__global__ void k_lin_g(const float* __restrict__ gf, const float* __restrict__ Wg,
                        const float* __restrict__ bg, float* __restrict__ out) {
  int i = blockIdx.x * blockDim.x + threadIdx.x;
  if (i >= GG * EMB) return;
  int g = i >> 7, c = i & (EMB - 1);
  float acc = bg[c];
#pragma unroll
  for (int k = 0; k < GFT; ++k) acc += gf[g * GFT + k] * Wg[k * EMB + c];
  out[i] = acc;
}

// ---- head: concat(mean,max,g_ft) @ Wo + bo -> log_softmax -----------------

__global__ void k_head(const float* __restrict__ sum, const unsigned* __restrict__ mx,
                       const float* __restrict__ cnt, const float* __restrict__ gft,
                       const float* __restrict__ Wo, const float* __restrict__ bo,
                       float* __restrict__ out) {
  int g = blockIdx.x * blockDim.x + threadIdx.x;
  if (g >= GG) return;
  float c = fmaxf(cnt[g], 1.0f);
  float l0 = bo[0], l1 = bo[1];
  for (int j = 0; j < 3 * EMB; ++j) {
    float v;
    if (j < EMB) v = sum[g * EMB + j] / c;
    else if (j < 2 * EMB) {
      float mv = decOrd(mx[g * EMB + (j - EMB)]);
      v = (mv > -1e30f) ? mv : 0.0f;           // empty segment -> 0
    } else v = gft[g * EMB + (j - 2 * EMB)];
    l0 += v * Wo[j * 2 + 0];
    l1 += v * Wo[j * 2 + 1];
  }
  float m = fmaxf(l0, l1);
  float lse = m + logf(__expf(l0 - m) + __expf(l1 - m));
  out[g * 2 + 0] = l0 - lse;
  out[g * 2 + 1] = l1 - lse;
}

// ---------------------------------------------------------------------------

static inline size_t rnd256(size_t x) { return (x + 255) & ~(size_t)255; }

extern "C" void kernel_launch(void* const* d_in, const int* in_sizes, int n_in,
                              void* d_out, int out_size, void* d_ws, size_t ws_size,
                              hipStream_t stream) {
  const float* x     = (const float*)d_in[0];
  const int*   edges = (const int*)d_in[1];
  const int*   bat   = (const int*)d_in[2];
  const float* gfeat = (const float*)d_in[3];
  const float* W1    = (const float*)d_in[4];
  const float* a1s   = (const float*)d_in[5];
  const float* a1d   = (const float*)d_in[6];
  const float* b1    = (const float*)d_in[7];
  const float* W2    = (const float*)d_in[8];
  const float* a2s   = (const float*)d_in[9];
  const float* a2d   = (const float*)d_in[10];
  const float* b2    = (const float*)d_in[11];
  const float* Wg    = (const float*)d_in[12];
  const float* bg    = (const float*)d_in[13];
  const float* Wo    = (const float*)d_in[14];
  const float* bo    = (const float*)d_in[15];
  float* out = (float*)d_out;

  const int* esrc = edges;
  const int* edst = edges + EE_RAW;

  // workspace carve-out (~98 MB total)
  char* w = (char*)d_ws;
  auto carve = [&](size_t bytes) { void* p = (void*)w; w += rnd256(bytes); return p; };
  unsigned short* xb   = (unsigned short*)carve((size_t)NN * F_IN * 2);
  unsigned short* w1t  = (unsigned short*)carve((size_t)EMB * F_IN * 2);
  unsigned short* w2t  = (unsigned short*)carve((size_t)EMB * EMB * 2);
  float*          H    = (float*)carve((size_t)NN * EMB * 4);   // GEMM out (per layer)
  float*          ss   = (float*)carve((size_t)NN * 4);
  float*          sd   = (float*)carve((size_t)NN * 4);
  unsigned*       menc = (unsigned*)carve((size_t)NN * 4);
  float*          den  = (float*)carve((size_t)NN * 4);
  float*          earr = (float*)carve((size_t)EE_ALL * 4);
  float*          parr = (float*)carve((size_t)EE_ALL * 4);
  float*          agg  = (float*)carve((size_t)NN * EMB * 4);   // layer output
  unsigned short* hb   = (unsigned short*)carve((size_t)NN * EMB * 2);
  float*          gsum = (float*)carve((size_t)GG * EMB * 4);
  unsigned*       gmax = (unsigned*)carve((size_t)GG * EMB * 4);
  float*          gcnt = (float*)carve((size_t)GG * 4);
  float*          gft  = (float*)carve((size_t)GG * EMB * 4);

  const int T = 256;
  auto blk = [](int n, int t) { return (n + t - 1) / t; };

  // ---- precision conversion ----
  k_f32_to_bf16<<<blk(NN * F_IN, T), T, 0, stream>>>(x, xb, NN * F_IN);
  k_transpose_bf16<<<blk(F_IN * EMB, T), T, 0, stream>>>(W1, w1t, F_IN, EMB);
  k_transpose_bf16<<<blk(EMB * EMB, T), T, 0, stream>>>(W2, w2t, EMB, EMB);

  for (int layer = 0; layer < 2; ++layer) {
    // ---- h = input @ W (WMMA bf16, fp32 accumulate) ----
    if (layer == 0)
      k_gemm_bf16_wmma<<<NN / 16, 256, 0, stream>>>(xb, w1t, H, NN, F_IN);
    else
      k_gemm_bf16_wmma<<<NN / 16, 256, 0, stream>>>(hb, w2t, H, NN, EMB);

    // ---- attention scalars per node ----
    k_node_dots<<<blk(NN * 32, T), T, 0, stream>>>(
        H, layer ? a2s : a1s, layer ? a2d : a1d, ss, sd, NN);

    // ---- init per-dst state + output accumulator ----
    k_fill_u32<<<blk(NN, T), T, 0, stream>>>(menc, ENC_NEG_INF, NN);
    k_fill_u32<<<blk(NN, T), T, 0, stream>>>((unsigned*)den, 0u, NN);
    k_fill_u32<<<blk(NN * EMB, T), T, 0, stream>>>((unsigned*)agg, 0u, NN * EMB);

    // ---- segment softmax + aggregation over (edges + self-loops) ----
    k_edge_logits<<<blk(EE_ALL, T), T, 0, stream>>>(esrc, edst, ss, sd, earr, menc);
    k_edge_exp<<<blk(EE_ALL, T), T, 0, stream>>>(edst, earr, menc, parr, den);
    k_edge_scatter<<<blk(EE_ALL * 32, T), T, 0, stream>>>(esrc, edst, parr, den, H, agg);

    // ---- + bias, and bf16 copy for next GEMM ----
    k_bias_bf16<<<blk(NN * EMB, T), T, 0, stream>>>(
        agg, layer ? b2 : b1, hb, NN * EMB);
  }

  // ---- global mean/max pooling ----
  k_fill_u32<<<blk(GG * EMB, T), T, 0, stream>>>((unsigned*)gsum, 0u, GG * EMB);
  k_fill_u32<<<blk(GG * EMB, T), T, 0, stream>>>(gmax, ENC_NEG_INF, GG * EMB);
  k_fill_u32<<<blk(GG, T), T, 0, stream>>>((unsigned*)gcnt, 0u, GG);
  k_pool<<<blk(NN * EMB, T), T, 0, stream>>>(agg, bat, gsum, gmax, gcnt);

  // ---- lin_g + head ----
  k_lin_g<<<blk(GG * EMB, T), T, 0, stream>>>(gfeat, Wg, bg, gft);
  k_head<<<blk(GG, T), T, 0, stream>>>(gsum, gmax, gcnt, gft, Wo, bo, out);
}
